// GCN_63617055588889
// MI455X (gfx1250) — matrix operand order repaired and checked by reading, not compiled, not run on hardware
//
#include <hip/hip_runtime.h>
#include <hip/hip_bf16.h>

// ---------------------------------------------------------------------------
// CDNA5 (gfx1250) GCN forward. Big GEMMs via V_WMMA_F32_16X16X32_BF16 with a
// software-pipelined (double-buffered) inner loop so loads for k-step n+1 are
// in flight while WMMAs for k-step n execute.
// ---------------------------------------------------------------------------

typedef __attribute__((ext_vector_type(16))) __bf16 v16bf;
typedef __attribute__((ext_vector_type(8)))  float  v8f;
typedef __attribute__((ext_vector_type(4)))  unsigned int v4u;

union Frag {
    v16bf v;
    v4u   q[2];
};

// ---------------------------- utility kernels ------------------------------

__global__ void k_zero(float* p, long n) {
    long i = (long)blockIdx.x * blockDim.x + threadIdx.x;
    if (i < n) p[i] = 0.0f;
}

__global__ void k_degree(const int* __restrict__ src, const int* __restrict__ dst,
                         float* odeg, float* ideg, int E) {
    int e = blockIdx.x * blockDim.x + threadIdx.x;
    if (e < E) {
        atomicAdd(&odeg[src[e]], 1.0f);
        atomicAdd(&ideg[dst[e]], 1.0f);
    }
}

__global__ void k_rsqrt_clamp(float* p, int n) {
    int i = blockIdx.x * blockDim.x + threadIdx.x;
    if (i < n) p[i] = rsqrtf(fmaxf(p[i], 1.0f));
}

__global__ void k_scatter0(const float* __restrict__ h, const int* __restrict__ src,
                           const int* __restrict__ dst, const float* __restrict__ onorm,
                           float* acc, int E) {
    int e = blockIdx.x * blockDim.x + threadIdx.x;
    if (e < E) {
        int s = src[e];
        atomicAdd(&acc[dst[e]], h[s] * onorm[s]);
    }
}

__global__ void k_mul_inorm(float* x, const float* __restrict__ inorm, int n) {
    int i = blockIdx.x * blockDim.x + threadIdx.x;
    if (i < n) x[i] *= inorm[i];
}

// Fused layer-1: msg[e,j] = relu(agg0s[src]*W1[j] + b1[j]) * out_norm[src],
// scattered into acc[dst, j].  D = 1024 hardcoded (256 vec4 chunks per edge).
__global__ void k_scatter_layer1(const float* __restrict__ agg0s, const float* __restrict__ W1,
                                 const float* __restrict__ b1, const int* __restrict__ src,
                                 const int* __restrict__ dst, const float* __restrict__ onorm,
                                 float* acc, int E) {
    long idx = (long)blockIdx.x * blockDim.x + threadIdx.x;
    long total = (long)E << 8;
    if (idx >= total) return;
    int e = (int)(idx >> 8);
    int j = ((int)idx & 255) << 2;
    int s = src[e], d = dst[e];
    float x = agg0s[s];
    float w = onorm[s];
    float4 wv = *reinterpret_cast<const float4*>(W1 + j);
    float4 bv = *reinterpret_cast<const float4*>(b1 + j);
    float* a = acc + ((size_t)d << 10) + j;
    atomicAdd(a + 0, fmaxf(x * wv.x + bv.x, 0.0f) * w);
    atomicAdd(a + 1, fmaxf(x * wv.y + bv.y, 0.0f) * w);
    atomicAdd(a + 2, fmaxf(x * wv.z + bv.z, 0.0f) * w);
    atomicAdd(a + 3, fmaxf(x * wv.w + bv.w, 0.0f) * w);
}

// Generic SpMM scatter: acc[dst, :] += x[src, :] * out_norm[src].
// chunks per edge = 1<<shiftC (vec4), D = 1<<shiftD.
__global__ void k_scatter(const float* __restrict__ x, const int* __restrict__ src,
                          const int* __restrict__ dst, const float* __restrict__ onorm,
                          float* acc, int E, int shiftC, int shiftD) {
    long idx = (long)blockIdx.x * blockDim.x + threadIdx.x;
    long total = (long)E << shiftC;
    if (idx >= total) return;
    int e = (int)(idx >> shiftC);
    int j = ((int)idx & ((1 << shiftC) - 1)) << 2;
    int s = src[e], d = dst[e];
    float w = onorm[s];
    float4 xv = *reinterpret_cast<const float4*>(x + ((size_t)s << shiftD) + j);
    float* a = acc + ((size_t)d << shiftD) + j;
    atomicAdd(a + 0, xv.x * w);
    atomicAdd(a + 1, xv.y * w);
    atomicAdd(a + 2, xv.z * w);
    atomicAdd(a + 3, xv.w * w);
}

// y[i] = bf16(x[i] * inorm[i >> shiftD])
__global__ void k_scale_bf16(const float* __restrict__ x, const float* __restrict__ inorm,
                             __bf16* __restrict__ y, long n, int shiftD) {
    long i = (long)blockIdx.x * blockDim.x + threadIdx.x;
    if (i < n) y[i] = (__bf16)(x[i] * inorm[i >> shiftD]);
}

// y[i] = relu(x[i] * inorm[i>>shiftD] + b[i & maskD])
__global__ void k_scale_bias_relu(const float* __restrict__ x, const float* __restrict__ inorm,
                                  const float* __restrict__ b, float* __restrict__ y,
                                  long n, int shiftD, int maskD) {
    long i = (long)blockIdx.x * blockDim.x + threadIdx.x;
    if (i < n) y[i] = fmaxf(x[i] * inorm[i >> shiftD] + b[(int)i & maskD], 0.0f);
}

// Wt[nc, k] = bf16(W[k, nc])   (W is [K, Nc] row-major)
__global__ void k_w_to_bf16_t(const float* __restrict__ W, __bf16* __restrict__ Wt,
                              int K, int Nc) {
    long idx = (long)blockIdx.x * blockDim.x + threadIdx.x;
    if (idx < (long)K * Nc) {
        int k  = (int)(idx / Nc);
        int nc = (int)(idx - (long)k * Nc);
        Wt[(size_t)nc * K + k] = (__bf16)W[idx];
    }
}

// ------------------------------ WMMA GEMM ----------------------------------
// C[M,Nc] = act(A[M,K] @ B[K,Nc] + bias), with Bt = B^T stored [Nc,K] bf16.
// One wave computes a 16x64 tile (4 N-subtiles reuse the A fragment).
// Double-buffered fragment pipeline: loads for k-step n+1 issued before the
// WMMAs for k-step n, so the compiler can wait on partial loadcnt instead of
// draining to zero before every matrix op.
// Requires K % 32 == 0, Nc % 64 == 0.  M handled with row guards.

__device__ __forceinline__ void load_afrag(Frag& f, const __bf16* aptr, int k0) {
    const v4u* aq = reinterpret_cast<const v4u*>(aptr + k0);
    f.q[0] = aq[0];  // K chunk [k0 + lhalf*8, +8)
    f.q[1] = aq[2];  // K chunk [k0 + 16 + lhalf*8, +8)
}

__device__ __forceinline__ void load_bfrag4(Frag* f, const __bf16* const* bptr, int k0) {
#pragma unroll
    for (int s = 0; s < 4; ++s) {
        const v4u* bq = reinterpret_cast<const v4u*>(bptr[s] + k0);
        f[s].q[0] = bq[0];
        f[s].q[1] = bq[2];
    }
}

__device__ __forceinline__ void wmma4(v8f* acc, const Frag& a, const Frag* b) {
#pragma unroll
    for (int s = 0; s < 4; ++s) {
        acc[s] = __builtin_amdgcn_wmma_f32_16x16x32_bf16(
            false, a.v, false, b[s].v, (short)0, acc[s], false, false);
    }
}

template <bool RELU, bool BIAS, typename OutT>
__global__ __launch_bounds__(256) void gemm_wmma(
    const __bf16* __restrict__ A,   // [M, K]  row-major
    const __bf16* __restrict__ Bt,  // [Nc, K] row-major (= B transposed)
    const float* __restrict__ bias, // [Nc] (unused if !BIAS)
    OutT* __restrict__ C,           // [M, Nc]
    int M, int K, int Nc) {
    const int lane  = threadIdx.x & 31;
    const int wave  = threadIdx.x >> 5;
    const int numM  = (M + 15) >> 4;
    const int numN  = Nc >> 6;
    const long unit = (long)blockIdx.x * (blockDim.x >> 5) + wave;
    if (unit >= (long)numM * numN) return;
    const int mt = (int)(unit / numN);
    const int nt = (int)(unit - (long)mt * numN);

    const int lmod  = lane & 15;
    const int lhalf = lane >> 4;

    // A fragment addressing (16-bit A 16x32 layout): lane m = lane&15,
    // elements 0..7  = A[m, k0 + lhalf*8 .. +7]
    // elements 8..15 = A[m, k0 + 16 + lhalf*8 .. +7]
    int arow = mt * 16 + lmod;
    if (arow >= M) arow = M - 1;  // clamp; masked on store
    const __bf16* aptr = A + (size_t)arow * K + lhalf * 8;

    // B fragment addressing (16-bit B 32x16): lane n = lane&15,
    // elements 0..7 = Bt[n, k0 + lhalf*8 .. +7], elements 8..15 = Bt[n, k0 + 16 + lhalf*8 .. +7]
    const int nbase = nt * 64;
    const __bf16* bptr[4];
#pragma unroll
    for (int s = 0; s < 4; ++s) {
        int ncol = nbase + s * 16 + lmod;
        bptr[s]  = Bt + (size_t)ncol * K + lhalf * 8;
    }

    v8f acc[4] = {};

    // Two-stage software pipeline (no dynamic register-array indexing).
    Frag a0, a1;
    Frag b0[4], b1[4];
    load_afrag(a0, aptr, 0);
    load_bfrag4(b0, bptr, 0);

    int k = 0;
    while (true) {
        int kn = k + 32;
        bool more = kn < K;
        if (more) {                       // prefetch into buffer 1
            load_afrag(a1, aptr, kn);
            load_bfrag4(b1, bptr, kn);
        }
        wmma4(acc, a0, b0);               // consume buffer 0
        if (!more) break;
        k = kn;

        kn = k + 32;
        more = kn < K;
        if (more) {                       // prefetch into buffer 0
            load_afrag(a0, aptr, kn);
            load_bfrag4(b0, bptr, kn);
        }
        wmma4(acc, a1, b1);               // consume buffer 1
        if (!more) break;
        k = kn;
    }

    // Epilogue: C layout lane l, vgpr r -> M = r + 8*(l>>4), N = l&15.
    const int rbase = mt * 16 + lhalf * 8;
#pragma unroll
    for (int s = 0; s < 4; ++s) {
        const int col = nbase + s * 16 + lmod;
        float bv = BIAS ? bias[col] : 0.0f;
#pragma unroll
        for (int r = 0; r < 8; ++r) {
            int row = rbase + r;
            if (row < M) {
                float v = acc[s][r] + bv;
                if (RELU) v = fmaxf(v, 0.0f);
                C[(size_t)row * Nc + col] = (OutT)v;
            }
        }
    }
}

// ------------------------------ pooling + MLP ------------------------------

__global__ void k_count(const int* __restrict__ gid, float* counts, int N) {
    int n = blockIdx.x * blockDim.x + threadIdx.x;
    if (n < N) atomicAdd(&counts[gid[n]], 1.0f);
}

__global__ void k_pool(const float* __restrict__ h, const int* __restrict__ gid,
                       float* hg, int N) {  // D = 512
    long idx = (long)blockIdx.x * blockDim.x + threadIdx.x;
    long total = (long)N << 9;
    if (idx < total) {
        int n = (int)(idx >> 9);
        int d = (int)idx & 511;
        atomicAdd(&hg[((size_t)gid[n] << 9) + d], h[idx]);
    }
}

__global__ void k_div_counts(float* hg, const float* __restrict__ counts, int G) {
    int idx = blockIdx.x * blockDim.x + threadIdx.x;
    if (idx < (G << 9)) hg[idx] /= fmaxf(counts[idx >> 9], 1.0f);
}

template <bool RELU>
__global__ void k_dense(const float* __restrict__ X, const float* __restrict__ W,
                        const float* __restrict__ b, float* __restrict__ Y,
                        int M, int K, int Nc) {
    int idx = blockIdx.x * blockDim.x + threadIdx.x;
    if (idx >= M * Nc) return;
    int i = idx / Nc, j = idx - i * Nc;
    float s = b[j];
    for (int k = 0; k < K; ++k) s += X[(size_t)i * K + k] * W[(size_t)k * Nc + j];
    if (RELU) s = fmaxf(s, 0.0f);
    Y[idx] = s;
}

// log_softmax over axis 0 (over the G graphs), logits are [G,2].
__global__ void k_logsoftmax_ax0(const float* __restrict__ logits, float* out, int G) {
    int j = threadIdx.x;
    if (j >= 2) return;
    float m = -__builtin_huge_valf();
    for (int i = 0; i < G; ++i) m = fmaxf(m, logits[i * 2 + j]);
    float s = 0.0f;
    for (int i = 0; i < G; ++i) s += expf(logits[i * 2 + j] - m);
    float lse = m + logf(s);
    for (int i = 0; i < G; ++i) out[i * 2 + j] = logits[i * 2 + j] - lse;
}

// ------------------------------- launcher ----------------------------------

extern "C" void kernel_launch(void* const* d_in, const int* in_sizes, int n_in,
                              void* d_out, int out_size, void* d_ws, size_t ws_size,
                              hipStream_t stream) {
    const int N = in_sizes[0];       // 10000
    const int E = in_sizes[1];       // 80000
    const int G = out_size / 2;      // 64

    const float* h   = (const float*)d_in[0];
    const int*   src = (const int*)d_in[1];
    const int*   dst = (const int*)d_in[2];
    const int*   gid = (const int*)d_in[3];
    const float* W1  = (const float*)d_in[4];
    const float* b1  = (const float*)d_in[5];
    const float* W2  = (const float*)d_in[6];
    const float* b2  = (const float*)d_in[7];
    const float* W3  = (const float*)d_in[8];
    const float* b3  = (const float*)d_in[9];
    const float* W4  = (const float*)d_in[10];
    const float* b4  = (const float*)d_in[11];
    const float* fw1 = (const float*)d_in[12];
    const float* fb1 = (const float*)d_in[13];
    const float* fw2 = (const float*)d_in[14];
    const float* fb2 = (const float*)d_in[15];
    const float* fw3 = (const float*)d_in[16];
    const float* fb3 = (const float*)d_in[17];
    float* out = (float*)d_out;

    // ---- workspace carve-up (deterministic bump allocator) ----
    char* ws = (char*)d_ws;
    size_t off = 0;
    auto alloc = [&](size_t bytes) -> char* {
        char* p = ws + off;
        off = (off + bytes + 255) & ~(size_t)255;
        return p;
    };
    float*  onorm  = (float*)alloc((size_t)N * 4);
    float*  inorm  = (float*)alloc((size_t)N * 4);
    float*  agg0   = (float*)alloc((size_t)N * 4);
    float*  counts = (float*)alloc((size_t)G * 4);
    float*  hg     = (float*)alloc((size_t)G * 512 * 4);
    float*  g1     = (float*)alloc((size_t)G * 512 * 4);
    float*  g2     = (float*)alloc((size_t)G * 256 * 4);
    float*  logits = (float*)alloc((size_t)G * 2 * 4);
    __bf16* W2t    = (__bf16*)alloc((size_t)1024 * 1024 * 2);
    __bf16* W3t    = (__bf16*)alloc((size_t)512 * 1024 * 2);
    __bf16* W4t    = (__bf16*)alloc((size_t)512 * 512 * 2);
    char*   bufA   = alloc((size_t)N * 1024 * 2);  // 20.5 MB rotating
    char*   bufB   = alloc((size_t)N * 1024 * 4);  // 41 MB rotating

    // rotating aliases (liveness-checked):
    float*  scat1 = (float*)bufB;   // [N,1024] f32
    __bf16* a1bf  = (__bf16*)bufA;  // [N,1024] bf16
    __bf16* h2bf  = (__bf16*)bufB;  // [N,1024] bf16 (over dead scat1)
    float*  a2pre = (float*)bufA;   // [N,512] f32   (over dead a1bf)
    float*  scat2 = (float*)bufB;   // [N,512] f32   (over dead h2bf)
    float*  h3    = (float*)bufA;   // [N,512] f32   (over dead a2pre)
    float*  scat3 = (float*)bufB;   // [N,512] f32   (over dead scat2)
    __bf16* a3bf  = (__bf16*)bufA;  // [N,512] bf16  (over dead h3)
    float*  h4    = (float*)bufB;   // [N,512] f32   (over dead scat3)

    const int B = 256;
    auto nb = [](long n, int b) { return (unsigned)((n + b - 1) / b); };

    // zero accumulators
    k_zero<<<nb(N, B), B, 0, stream>>>(onorm, N);
    k_zero<<<nb(N, B), B, 0, stream>>>(inorm, N);
    k_zero<<<nb(N, B), B, 0, stream>>>(agg0, N);
    k_zero<<<nb(G, B), B, 0, stream>>>(counts, G);
    k_zero<<<nb((long)G * 512, B), B, 0, stream>>>(hg, (long)G * 512);

    // bf16-transposed weights (independent of graph work)
    k_w_to_bf16_t<<<nb((long)1024 * 1024, B), B, 0, stream>>>(W2, W2t, 1024, 1024);
    k_w_to_bf16_t<<<nb((long)1024 * 512, B), B, 0, stream>>>(W3, W3t, 1024, 512);
    k_w_to_bf16_t<<<nb((long)512 * 512, B), B, 0, stream>>>(W4, W4t, 512, 512);

    // degrees -> norms
    k_degree<<<nb(E, B), B, 0, stream>>>(src, dst, onorm, inorm, E);
    k_rsqrt_clamp<<<nb(N, B), B, 0, stream>>>(onorm, N);
    k_rsqrt_clamp<<<nb(N, B), B, 0, stream>>>(inorm, N);

    // layer 0 aggregate: agg0 = in_norm * scatter(h * out_norm)
    k_scatter0<<<nb(E, B), B, 0, stream>>>(h, src, dst, onorm, agg0, E);
    k_mul_inorm<<<nb(N, B), B, 0, stream>>>(agg0, inorm, N);

    // layer 1 (1->1024), fused into the scatter; then scale+bf16 -> GEMM2 A
    k_zero<<<nb((long)N * 1024, B), B, 0, stream>>>(scat1, (long)N * 1024);
    k_scatter_layer1<<<nb((long)E * 256, B), B, 0, stream>>>(agg0, W1, b1, src, dst, onorm,
                                                             scat1, E);
    k_scale_bf16<<<nb((long)N * 1024, B), B, 0, stream>>>(scat1, inorm, a1bf,
                                                          (long)N * 1024, 10);

    // GEMM2: h2 = relu(a1 @ W2 + b2)  -> bf16
    {
        long units = (long)((N + 15) >> 4) * (1024 >> 6);
        gemm_wmma<true, true, __bf16><<<nb(units, 8), 256, 0, stream>>>(
            a1bf, W2t, b2, h2bf, N, 1024, 1024);
    }
    // GEMM3: a2pre = h2 @ W3 (no bias/relu) -> f32
    {
        long units = (long)((N + 15) >> 4) * (512 >> 6);
        gemm_wmma<false, false, float><<<nb(units, 8), 256, 0, stream>>>(
            h2bf, W3t, nullptr, a2pre, N, 1024, 512);
    }

    // aggregate(a2pre) -> h3 = relu(. * in_norm + b3)
    k_zero<<<nb((long)N * 512, B), B, 0, stream>>>(scat2, (long)N * 512);
    k_scatter<<<nb((long)E * 128, B), B, 0, stream>>>(a2pre, src, dst, onorm, scat2, E, 7, 9);
    k_scale_bias_relu<<<nb((long)N * 512, B), B, 0, stream>>>(scat2, inorm, b3, h3,
                                                              (long)N * 512, 9, 511);

    // aggregate(h3) -> a3 (bf16)
    k_zero<<<nb((long)N * 512, B), B, 0, stream>>>(scat3, (long)N * 512);
    k_scatter<<<nb((long)E * 128, B), B, 0, stream>>>(h3, src, dst, onorm, scat3, E, 7, 9);
    k_scale_bf16<<<nb((long)N * 512, B), B, 0, stream>>>(scat3, inorm, a3bf,
                                                         (long)N * 512, 9);

    // GEMM4: h4 = relu(a3 @ W4 + b4) -> f32
    {
        long units = (long)((N + 15) >> 4) * (512 >> 6);
        gemm_wmma<true, true, float><<<nb(units, 8), 256, 0, stream>>>(
            a3bf, W4t, b4, h4, N, 512, 512);
    }

    // mean_nodes pooling
    k_count<<<nb(N, B), B, 0, stream>>>(gid, counts, N);
    k_pool<<<nb((long)N * 512, B), B, 0, stream>>>(h4, gid, hg, N);
    k_div_counts<<<nb((long)G * 512, B), B, 0, stream>>>(hg, counts, G);

    // small MLP head (fp32)
    k_dense<true><<<nb((long)G * 512, B), B, 0, stream>>>(hg, fw1, fb1, g1, G, 512, 512);
    k_dense<true><<<nb((long)G * 256, B), B, 0, stream>>>(g1, fw2, fb2, g2, G, 512, 256);
    k_dense<false><<<nb((long)G * 2, B), B, 0, stream>>>(g2, fw3, fb3, logits, G, 256, 2);

    // log_softmax over graphs (axis 0)
    k_logsoftmax_ax0<<<1, 64, 0, stream>>>(logits, out, G);
}
